// Net_89764816486721
// MI455X (gfx1250) — matrix operand order repaired and checked by reading, not compiled, not run on hardware
//
#include <hip/hip_runtime.h>

#define B_ 4
#define T_ 300
#define U_ 64
#define H_ 512
#define V_ 1024

typedef __attribute__((ext_vector_type(16))) _Float16 v16h;
typedef __attribute__((ext_vector_type(4)))  _Float16 v4h;
typedef __attribute__((ext_vector_type(8)))  float    v8f;
typedef __attribute__((ext_vector_type(2)))  float    v2f;

// Fast activations: CDNA5 has a native TANH transcendental; fall back to a
// 4-op exp/rcp form (saturates correctly at +-inf). Sigmoid = rcp(1+exp(-x)).
static __device__ __forceinline__ float fast_tanh(float v) {
#if __has_builtin(__builtin_amdgcn_tanhf)
    return __builtin_amdgcn_tanhf(v);
#else
    float e = __expf(2.0f * v);
    return 1.0f - __fdividef(2.0f, e + 1.0f);
#endif
}
static __device__ __forceinline__ float fast_sigmoid(float v) {
    return __fdividef(1.0f, 1.0f + __expf(-v));
}

// ---------------------------------------------------------------------------
// Kernel 1: first-layer projections in full fp32 via V_WMMA_F32_16X16X4_F32.
// Each block: 16 rows x 64 cols of BOTH the W1-projection and the Wg-projection
// (shared A fragments). Part 0 handles x rows (hx, gx, no bias); part 1 handles
// y rows (hyb = y@W1[H:]+b1, gyb = y@Wg[H:]+bg  -- biases folded here).
// ---------------------------------------------------------------------------
__global__ __launch_bounds__(128)
void proj_kernel(const float* __restrict__ x, const float* __restrict__ y,
                 const float* __restrict__ W1, const float* __restrict__ b1,
                 const float* __restrict__ Wg, const float* __restrict__ bg,
                 float* __restrict__ hx, float* __restrict__ gx,
                 float* __restrict__ hyb, float* __restrict__ gyb)
{
    __shared__ float lds_in[16][H_];

    const int NBX = (1200 / 16) * (H_ / 64);   // 600 blocks for the x part
    int bid = blockIdx.x;
    const float* src;
    float *dst1, *dstg;
    int rowTile, colTile, wOff;
    const float* bias1 = nullptr;
    const float* biasg = nullptr;
    if (bid < NBX) {
        rowTile = bid / (H_ / 64);
        colTile = bid % (H_ / 64);
        src = x; dst1 = hx; dstg = gx; wOff = 0;
    } else {
        int b2id = bid - NBX;
        rowTile = b2id / (H_ / 64);
        colTile = b2id % (H_ / 64);
        src = y; dst1 = hyb; dstg = gyb; wOff = H_;
        bias1 = b1; biasg = bg;
    }
    const int row0 = rowTile * 16;

    for (int idx = threadIdx.x; idx < 16 * H_; idx += 128) {
        int m = idx >> 9;
        int k = idx & (H_ - 1);
        lds_in[m][k] = src[(size_t)(row0 + m) * H_ + k];
    }
    __syncthreads();

    const int lane = threadIdx.x & 31;
    const int wave = threadIdx.x >> 5;
    const int n    = colTile * 64 + wave * 16 + (lane & 15);
    const int kg   = lane >> 4;
    const int mrow = lane & 15;

    v8f acc1 = {};
    v8f accg = {};
    for (int k0 = 0; k0 < H_; k0 += 4) {
        int ka = k0 + 2 * kg;            // 32-bit A/B layout: element j -> K = 2*kg + j
        v2f a, bw1, bwg;
        a.x   = lds_in[mrow][ka];
        a.y   = lds_in[mrow][ka + 1];
        bw1.x = W1[(size_t)(wOff + ka) * H_ + n];
        bw1.y = W1[(size_t)(wOff + ka + 1) * H_ + n];
        bwg.x = Wg[(size_t)(wOff + ka) * H_ + n];
        bwg.y = Wg[(size_t)(wOff + ka + 1) * H_ + n];
        acc1 = __builtin_amdgcn_wmma_f32_16x16x4_f32(false, a, false, bw1, (short)0, acc1, false, false);
        accg = __builtin_amdgcn_wmma_f32_16x16x4_f32(false, a, false, bwg, (short)0, accg, false, false);
    }

    const float add1 = bias1 ? bias1[n] : 0.0f;
    const float addg = biasg ? biasg[n] : 0.0f;
#pragma unroll
    for (int j = 0; j < 8; ++j) {
        int m = j + 8 * kg;              // C layout: VGPR j -> M = j + 8*(lane>=16)
        dst1[(size_t)(row0 + m) * H_ + n] = acc1[j] + add1;
        dstg[(size_t)(row0 + m) * H_ + n] = accg[j] + addg;
    }
}

// ---------------------------------------------------------------------------
// Kernel 2: repack W2 (H x V fp32, row-major) into f16 WMMA B-fragment order:
//   w2h[ ((ks*64 + nt)*32 + lane)*16 + j ] = W2[k*V + n]  (f16)
// with  n = nt*16 + (lane&15),  k = ks*32 + j + 16*(lane>>4)
// so each lane's 16 halves for one (ks,nt) fragment are contiguous 32 bytes.
// ---------------------------------------------------------------------------
__global__ __launch_bounds__(256)
void packW2_kernel(const float* __restrict__ W2, _Float16* __restrict__ w2h)
{
    int linear = blockIdx.x * 256 + threadIdx.x;   // 0 .. 32767  (= 16ks * 64nt * 32lane)
    int lane = linear & 31;
    int nt   = (linear >> 5) & 63;
    int ks   = linear >> 11;
    int n    = nt * 16 + (lane & 15);
    int kg   = lane >> 4;
    _Float16* dst = w2h + (size_t)linear * 16;
#pragma unroll
    for (int j = 0; j < 16; ++j) {
        int k = ks * 32 + j + 16 * kg;
        dst[j] = (_Float16)W2[(size_t)k * V_ + n];
    }
}

// ---------------------------------------------------------------------------
// Kernel 3: fused joint network, M=64 rows per block (one (b,t), all U=64),
// 512 threads = 16 waves (4 waves/SIMD for latency hiding at 1 block/WGP).
// nt-outer loop (4 N-tiles per wave): each B fragment (2x global_load_b128
// from L2) feeds 4 M-tile WMMAs -> W2 L2 traffic amortized to 1 MB / 64 rows.
// LDS: A frags 64 KB + fp32 out tile 256 KB = 320 KB (CDNA5 per-WG max).
// Epilogue: +b2, fused row-wise log_softmax, vectorized b128 stores.
// ---------------------------------------------------------------------------
__global__ __launch_bounds__(512)
void joint_kernel(const float* __restrict__ hx, const float* __restrict__ gx,
                  const float* __restrict__ hyb, const float* __restrict__ gyb,
                  const _Float16* __restrict__ w2h, const float* __restrict__ b2,
                  float* __restrict__ out)
{
    extern __shared__ __align__(32) char smem[];
    _Float16* lds_a   = (_Float16*)smem;                 // [4][16][32][16] halves = 64 KB
    float*    lds_out = (float*)(smem + 64 * 1024);      // [64][1024] fp32 = 256 KB

    const int bt = blockIdx.x;           // 0 .. B*T-1
    const int b  = bt / T_;
    const int rowbase = bt * U_;         // 64 rows, u = 0..63

    const float4* hxr4 = (const float4*)(hx + (size_t)bt * H_);
    const float4* gxr4 = (const float4*)(gx + (size_t)bt * H_);
    const float4* hyb4 = (const float4*)(hyb + (size_t)(b * U_) * H_);
    const float4* gyb4 = (const float4*)(gyb + (size_t)(b * U_) * H_);

    const int tid = threadIdx.x;

    // ---- stage activations as f16 A-fragments (contiguous k = 16*kg + j order, same as w2h) ----
    // 64 rows x 128 float4 = 8192 float4; 16 iterations per thread.
    for (int idx = tid; idx < 64 * (H_ / 4); idx += 512) {
        int m = idx >> 7;                 // row in tile (u = m)
        int q = idx & 127;                // float4 index within row
        int k = q * 4;
        float4 hv = hxr4[q];
        float4 gv = gxr4[q];
        float4 hy = hyb4[(size_t)m * (H_ / 4) + q];
        float4 gy = gyb4[(size_t)m * (H_ / 4) + q];
        float a0 = fast_tanh(hv.x + hy.x) * fast_sigmoid(gv.x + gy.x);
        float a1 = fast_tanh(hv.y + hy.y) * fast_sigmoid(gv.y + gy.y);
        float a2 = fast_tanh(hv.z + hy.z) * fast_sigmoid(gv.z + gy.z);
        float a3 = fast_tanh(hv.w + hy.w) * fast_sigmoid(gv.w + gy.w);
        // fragment mapping for k..k+3 (same lane, contiguous j since k%4==0)
        int ks = k >> 5;
        int kk = k & 31;
        int kg = kk >> 4;
        int j0 = kk & 15;
        int lane_s = (m & 15) | (kg << 4);
        int mt = m >> 4;
        v4h pk; pk.x = (_Float16)a0; pk.y = (_Float16)a1; pk.z = (_Float16)a2; pk.w = (_Float16)a3;
        *(v4h*)(lds_a + ((size_t)((mt * 16 + ks) * 32 + lane_s) * 16) + j0) = pk;
    }
    __syncthreads();

    const int lane = tid & 31;
    const int wave = tid >> 5;            // 0..15
    const v16h* w2v = (const v16h*)w2h;
    const v16h* av  = (const v16h*)lds_a;

    for (int nti = 0; nti < 4; ++nti) {
        const int nt = wave * 4 + nti;    // 16 waves x 4 = 64 N-tiles
        v8f acc[4] = {{}, {}, {}, {}};
#pragma unroll 4
        for (int ks = 0; ks < 16; ++ks) {
            v16h bf = w2v[(size_t)(ks * 64 + nt) * 32 + lane];  // 2x global_load_b128 (L2)
#pragma unroll
            for (int mt = 0; mt < 4; ++mt) {
                v16h am = av[(size_t)(mt * 16 + ks) * 32 + lane];
                acc[mt] = __builtin_amdgcn_wmma_f32_16x16x32_f16(false, am, false, bf, (short)0, acc[mt], false, false);
            }
        }
        const int n  = nt * 16 + (lane & 15);
        const int kg = lane >> 4;
        const float bias = b2[n];
#pragma unroll
        for (int mt = 0; mt < 4; ++mt) {
#pragma unroll
            for (int j = 0; j < 8; ++j) {
                int m = mt * 16 + j + 8 * kg;
                lds_out[(size_t)m * V_ + n] = acc[mt][j] + bias;
            }
        }
    }
    __syncthreads();

    // ---- fused log_softmax over V (vectorized), 4 rows per wave ----
    for (int ri = 0; ri < 4; ++ri) {
        const int r = wave * 4 + ri;
        const float4* rowp4 = (const float4*)(lds_out + (size_t)r * V_);
        float mx = -INFINITY;
        for (int q = lane; q < V_ / 4; q += 32) {
            float4 vv = rowp4[q];
            mx = fmaxf(mx, fmaxf(fmaxf(vv.x, vv.y), fmaxf(vv.z, vv.w)));
        }
#pragma unroll
        for (int s = 16; s >= 1; s >>= 1) mx = fmaxf(mx, __shfl_xor(mx, s, 32));
        float sum = 0.0f;
        for (int q = lane; q < V_ / 4; q += 32) {
            float4 vv = rowp4[q];
            sum += __expf(vv.x - mx) + __expf(vv.y - mx) + __expf(vv.z - mx) + __expf(vv.w - mx);
        }
#pragma unroll
        for (int s = 16; s >= 1; s >>= 1) sum += __shfl_xor(sum, s, 32);
        const float lse = mx + __logf(sum);
        float4* op4 = (float4*)(out + (size_t)(rowbase + r) * V_);
        for (int q = lane; q < V_ / 4; q += 32) {
            float4 vv = rowp4[q];
            float4 ov; ov.x = vv.x - lse; ov.y = vv.y - lse; ov.z = vv.z - lse; ov.w = vv.w - lse;
            op4[q] = ov;                                   // global_store_b128
        }
    }
}

// ---------------------------------------------------------------------------
extern "C" void kernel_launch(void* const* d_in, const int* in_sizes, int n_in,
                              void* d_out, int out_size, void* d_ws, size_t ws_size,
                              hipStream_t stream) {
    (void)in_sizes; (void)n_in; (void)out_size; (void)ws_size;
    const float* x  = (const float*)d_in[0];
    const float* y  = (const float*)d_in[1];
    const float* W1 = (const float*)d_in[2];
    const float* b1 = (const float*)d_in[3];
    const float* Wg = (const float*)d_in[4];
    const float* bg = (const float*)d_in[5];
    const float* W2 = (const float*)d_in[6];
    const float* b2 = (const float*)d_in[7];
    float* out = (float*)d_out;

    // workspace layout
    float* hx  = (float*)d_ws;                          // 1200*512
    float* gx  = hx  + (size_t)1200 * 512;              // 1200*512
    float* hyb = gx  + (size_t)1200 * 512;              // 256*512
    float* gyb = hyb + (size_t)256 * 512;               // 256*512
    _Float16* w2h = (_Float16*)(gyb + (size_t)256 * 512);  // 512*1024 halves (32B-aligned)

    // 1) projections (600 blocks for x-part + 128 for y-part)
    proj_kernel<<<dim3(600 + 128), dim3(128), 0, stream>>>(
        x, y, W1, b1, Wg, bg, hx, gx, hyb, gyb);

    // 2) pack W2 into f16 B-fragment layout
    packW2_kernel<<<dim3(128), dim3(256), 0, stream>>>(W2, w2h);

    // 3) fused joint + log_softmax: B*T = 1200 blocks, 320 KB dynamic LDS
    const int smem_bytes = 64 * 1024 + 64 * V_ * (int)sizeof(float);  // 327680
    hipFuncSetAttribute((const void*)joint_kernel,
                        hipFuncAttributeMaxDynamicSharedMemorySize, smem_bytes);
    joint_kernel<<<dim3(1200), dim3(512), smem_bytes, stream>>>(
        hx, gx, hyb, gyb, w2h, b2, out);
}
